// RahmenGraph_69956427317270
// MI455X (gfx1250) — matrix operand order repaired and checked by reading, not compiled, not run on hardware
//
#include <hip/hip_runtime.h>
#include <hip/hip_fp16.h>

typedef __attribute__((ext_vector_type(16))) _Float16 v16h;
typedef __attribute__((ext_vector_type(4)))  _Float16 v4h;
typedef __attribute__((ext_vector_type(8)))  float    v8f;

#define N_NODES 50000
#define N_EDGES 640000
#define N_REL   4
#define F_DIM   128
#define A_DIM   64

// ---------------------------------------------------------------- zero
__global__ void zero_f32(float* __restrict__ p, long n) {
  long i = (long)blockIdx.x * blockDim.x + threadIdx.x;
  if (i < n) p[i] = 0.0f;
}

// ------------------------------------------- weight convert + transpose
// src: [R, K, Nn] f32  ->  dst: [R, Nn, K] f16
__global__ void cvt_transpose(const float* __restrict__ src, _Float16* __restrict__ dst,
                              int K, int Nn) {
  long i = (long)blockIdx.x * blockDim.x + threadIdx.x;
  long tot = (long)N_REL * K * Nn;
  if (i >= tot) return;
  int k = (int)(i % K);
  long t = i / K;
  int nn = (int)(t % Nn);
  int r  = (int)(t / Nn);
  dst[((long)r * Nn + nn) * K + k] = (_Float16)src[((long)r * K + k) * Nn + nn];
}

// ---------------------------------------------------------- edge scatter
// one wave (32 lanes) per (relation, edge); lane covers 4 floats of the row
__global__ __launch_bounds__(256)
void edge_scatter(const float* __restrict__ feat,
                  const int* __restrict__ srcI,
                  const int* __restrict__ dstI,
                  float* __restrict__ sum, float* __restrict__ cnt) {
  long wid = (long)blockIdx.x * 8 + threadIdx.y;
  if (wid >= (long)N_REL * N_EDGES) return;
  int r = (int)(wid / N_EDGES);
  int e = (int)(wid % N_EDGES);
  int s = srcI[(long)r * N_EDGES + e];
  int d = dstI[(long)r * N_EDGES + e];
  int lane = threadIdx.x;
  float4 v = *(const float4*)(feat + (long)s * F_DIM + lane * 4);
  float* dp = sum + ((long)r * N_NODES + d) * F_DIM + lane * 4;
  __hip_atomic_fetch_add(dp + 0, v.x, __ATOMIC_RELAXED, __HIP_MEMORY_SCOPE_AGENT);
  __hip_atomic_fetch_add(dp + 1, v.y, __ATOMIC_RELAXED, __HIP_MEMORY_SCOPE_AGENT);
  __hip_atomic_fetch_add(dp + 2, v.z, __ATOMIC_RELAXED, __HIP_MEMORY_SCOPE_AGENT);
  __hip_atomic_fetch_add(dp + 3, v.w, __ATOMIC_RELAXED, __HIP_MEMORY_SCOPE_AGENT);
  if (lane == 0)
    __hip_atomic_fetch_add(cnt + (long)r * N_NODES + d, 1.0f,
                           __ATOMIC_RELAXED, __HIP_MEMORY_SCOPE_AGENT);
}

// --------------------------------------------------- h_rel = feat + mean
__global__ void build_hrel(const float* __restrict__ feat, const float* __restrict__ sum,
                           const float* __restrict__ cnt, _Float16* __restrict__ hrel) {
  long i = (long)blockIdx.x * blockDim.x + threadIdx.x;   // float4 chunks
  long tot = (long)N_REL * N_NODES * (F_DIM / 4);
  if (i >= tot) return;
  int c = (int)(i % (F_DIM / 4));
  long t = i / (F_DIM / 4);
  int n = (int)(t % N_NODES);
  int r = (int)(t / N_NODES);
  float4 f4 = *(const float4*)(feat + (long)n * F_DIM + c * 4);
  float4 s4 = *(const float4*)(sum + ((long)r * N_NODES + n) * F_DIM + c * 4);
  float cc  = cnt[(long)r * N_NODES + n];
  float inv = cc > 0.0f ? 1.0f / cc : 0.0f;
  v4h o;
  o.x = (_Float16)(f4.x + s4.x * inv);
  o.y = (_Float16)(f4.y + s4.y * inv);
  o.z = (_Float16)(f4.z + s4.z * inv);
  o.w = (_Float16)(f4.w + s4.w * inv);
  *(v4h*)(hrel + ((long)r * N_NODES + n) * F_DIM + c * 4) = o;
}

// ------------------------------------------------ WMMA GEMM  [16 x 128]
// C[16x128] = A[16x128] * Wt^T + bias  (Wt stored [R][out=128][in=128] f16)
// one wave per (r, 16-node tile); v_wmma_f32_16x16x32_f16, 8 N-tiles x 4 K-steps
__global__ __launch_bounds__(256)
void gemm128(const _Float16* __restrict__ A, const _Float16* __restrict__ Wt,
             const float* __restrict__ bias,
             _Float16* __restrict__ out16, float* __restrict__ out32) {
  long wid = (long)blockIdx.x * 8 + threadIdx.y;
  const long MT = (long)N_REL * (N_NODES / 16);
  if (wid >= MT) return;
  int r     = (int)(wid / (N_NODES / 16));
  int mbase = (int)(wid % (N_NODES / 16)) * 16;
  int lane = threadIdx.x;
  int lm = lane & 15, hi = lane >> 4;
  const _Float16* Arow = A + ((long)r * N_NODES + mbase + lm) * F_DIM;

  v8f c[8] = {};
  for (int ks = 0; ks < 4; ks++) {
    int kb = ks * 32;
    // 16-bit A 16x32 fragment: lane<16 -> K = kb+{0..7, 16..23}; lane>=16 -> +8
    v16h a;
    ((float4*)&a)[0] = *(const float4*)(Arow + kb + hi * 8);
    ((float4*)&a)[1] = *(const float4*)(Arow + kb + hi * 8 + 16);
#pragma unroll
    for (int nt = 0; nt < 8; nt++) {
      // B 32x16 fragment: lane n=lm, K = kb + hi*8 + {0..7, 16..23} (same order as A)
      const _Float16* wrow = Wt + ((long)r * F_DIM + nt * 16 + lm) * F_DIM + kb + hi * 8;
      v16h b;
      ((float4*)&b)[0] = *(const float4*)(wrow);
      ((float4*)&b)[1] = *(const float4*)(wrow + 16);
      c[nt] = __builtin_amdgcn_wmma_f32_16x16x32_f16(false, a, false, b,
                                                     (short)0, c[nt], false, false);
    }
  }
#pragma unroll
  for (int nt = 0; nt < 8; nt++) {
    float bv = bias[r * F_DIM + nt * 16 + lm];
#pragma unroll
    for (int j = 0; j < 8; j++) {
      int row = mbase + j + hi * 8;                 // C layout: VGPR j -> M=j (+8 for hi)
      long idx = ((long)r * N_NODES + row) * F_DIM + nt * 16 + lm;
      float val = c[nt][j] + bv;
      if (out32) out32[idx] = val;
      if (out16) out16[idx] = (_Float16)val;
    }
  }
}

// --------------------------------- semantic attention logits via WMMA
// logit[r][n] = sum_a tanh( (h16 @ Ws1)[n][a] ) * ws2[a]
__global__ __launch_bounds__(256)
void attn_logits(const _Float16* __restrict__ h16, const _Float16* __restrict__ Ws1t,
                 const float* __restrict__ ws2, float* __restrict__ logit) {
  long wid = (long)blockIdx.x * 8 + threadIdx.y;
  const long MT = (long)N_REL * (N_NODES / 16);
  if (wid >= MT) return;
  int r     = (int)(wid / (N_NODES / 16));
  int mbase = (int)(wid % (N_NODES / 16)) * 16;
  int lane = threadIdx.x;
  int lm = lane & 15, hi = lane >> 4;
  const _Float16* Arow = h16 + ((long)r * N_NODES + mbase + lm) * F_DIM;

  v16h a[4];
#pragma unroll
  for (int ks = 0; ks < 4; ks++) {
    int kb = ks * 32;
    ((float4*)&a[ks])[0] = *(const float4*)(Arow + kb + hi * 8);
    ((float4*)&a[ks])[1] = *(const float4*)(Arow + kb + hi * 8 + 16);
  }

  float pj[8] = {0.f, 0.f, 0.f, 0.f, 0.f, 0.f, 0.f, 0.f};
#pragma unroll
  for (int nt = 0; nt < 4; nt++) {                    // A_DIM = 64 -> 4 tiles
    v8f c = {};
#pragma unroll
    for (int ks = 0; ks < 4; ks++) {
      int kb = ks * 32;
      const _Float16* wrow = Ws1t + ((long)r * A_DIM + nt * 16 + lm) * F_DIM + kb + hi * 8;
      v16h b;
      ((float4*)&b)[0] = *(const float4*)(wrow);
      ((float4*)&b)[1] = *(const float4*)(wrow + 16);
      c = __builtin_amdgcn_wmma_f32_16x16x32_f16(false, a[ks], false, b,
                                                 (short)0, c, false, false);
    }
    float s2 = ws2[r * A_DIM + nt * 16 + lm];
#pragma unroll
    for (int j = 0; j < 8; j++) pj[j] += tanhf(c[j]) * s2;
  }
  // reduce over the 16-lane N groups (lanes 0-15 hold M=j, lanes 16-31 hold M=j+8)
#pragma unroll
  for (int j = 0; j < 8; j++) {
    for (int off = 8; off; off >>= 1) pj[j] += __shfl_xor(pj[j], off, 16);
  }
  if (lm == 0) {
#pragma unroll
    for (int j = 0; j < 8; j++)
      logit[(long)r * N_NODES + mbase + j + hi * 8] = pj[j];
  }
}

// -------------------------- softmax over relations + weighted sum + reduce
__global__ __launch_bounds__(128)
void finalize(const float* __restrict__ h, const float* __restrict__ logit,
              float* __restrict__ out) {
  int d  = threadIdx.x;                 // 128
  int n0 = blockIdx.x * 64;
  int nend = n0 + 64; if (nend > N_NODES) nend = N_NODES;
  float acc = 0.0f;
  for (int n = n0; n < nend; n++) {
    float l0 = logit[n];
    float l1 = logit[N_NODES + n];
    float l2 = logit[2L * N_NODES + n];
    float l3 = logit[3L * N_NODES + n];
    float m = fmaxf(fmaxf(l0, l1), fmaxf(l2, l3));
    float e0 = __expf(l0 - m), e1 = __expf(l1 - m);
    float e2 = __expf(l2 - m), e3 = __expf(l3 - m);
    float inv = 1.0f / (e0 + e1 + e2 + e3);
    acc += inv * (e0 * h[((long)0 * N_NODES + n) * F_DIM + d] +
                  e1 * h[((long)1 * N_NODES + n) * F_DIM + d] +
                  e2 * h[((long)2 * N_NODES + n) * F_DIM + d] +
                  e3 * h[((long)3 * N_NODES + n) * F_DIM + d]);
  }
  __hip_atomic_fetch_add(&out[d], acc, __ATOMIC_RELAXED, __HIP_MEMORY_SCOPE_AGENT);
}

// ---------------------------------------------------------------- launch
extern "C" void kernel_launch(void* const* d_in, const int* in_sizes, int n_in,
                              void* d_out, int out_size, void* d_ws, size_t ws_size,
                              hipStream_t stream) {
  const float* feat = (const float*)d_in[0];
  const int*   srcI = (const int*)d_in[1];
  const int*   dstI = (const int*)d_in[2];
  const float* W1   = (const float*)d_in[3];
  const float* b1   = (const float*)d_in[4];
  const float* W2   = (const float*)d_in[5];
  const float* b2   = (const float*)d_in[6];
  const float* Ws1  = (const float*)d_in[7];
  const float* Ws2  = (const float*)d_in[8];
  float* out = (float*)d_out;

  char* ws = (char*)d_ws;
  size_t off = 0;
  auto carve = [&](size_t bytes) -> void* {
    off = (off + 255) & ~(size_t)255;
    void* p = ws + off;
    off += bytes;
    return p;
  };
  float*    sum   = (float*)carve((size_t)N_REL * N_NODES * F_DIM * 4); // aliased -> h32
  float*    cnt   = (float*)carve((size_t)N_REL * N_NODES * 4);
  _Float16* hrel  = (_Float16*)carve((size_t)N_REL * N_NODES * F_DIM * 2); // aliased -> h16
  _Float16* h1    = (_Float16*)carve((size_t)N_REL * N_NODES * F_DIM * 2);
  float*    logit = (float*)carve((size_t)N_REL * N_NODES * 4);
  _Float16* W1t   = (_Float16*)carve((size_t)N_REL * F_DIM * F_DIM * 2);
  _Float16* W2t   = (_Float16*)carve((size_t)N_REL * F_DIM * F_DIM * 2);
  _Float16* Ws1t  = (_Float16*)carve((size_t)N_REL * A_DIM * F_DIM * 2);
  float*    h32 = sum;   // sum dead after build_hrel
  _Float16* h16 = hrel;  // hrel dead after gemm1

  long nz = (long)N_REL * N_NODES * F_DIM;
  zero_f32<<<(int)((nz + 255) / 256), 256, 0, stream>>>(sum, nz);
  long nc = (long)N_REL * N_NODES;
  zero_f32<<<(int)((nc + 255) / 256), 256, 0, stream>>>(cnt, nc);
  zero_f32<<<1, 128, 0, stream>>>(out, 128);

  long tw = (long)N_REL * F_DIM * F_DIM;
  cvt_transpose<<<(int)((tw + 255) / 256), 256, 0, stream>>>(W1, W1t, F_DIM, F_DIM);
  cvt_transpose<<<(int)((tw + 255) / 256), 256, 0, stream>>>(W2, W2t, F_DIM, F_DIM);
  long ta = (long)N_REL * F_DIM * A_DIM;
  cvt_transpose<<<(int)((ta + 255) / 256), 256, 0, stream>>>(Ws1, Ws1t, F_DIM, A_DIM);

  dim3 wgrp(32, 8);
  long nw = (long)N_REL * N_EDGES;
  edge_scatter<<<(int)((nw + 7) / 8), wgrp, 0, stream>>>(feat, srcI, dstI, sum, cnt);

  long nh = (long)N_REL * N_NODES * (F_DIM / 4);
  build_hrel<<<(int)((nh + 255) / 256), 256, 0, stream>>>(feat, sum, cnt, hrel);

  long ngw = (long)N_REL * (N_NODES / 16);       // 12500 waves
  gemm128<<<(int)((ngw + 7) / 8), wgrp, 0, stream>>>(hrel, W1t, b1, h1, nullptr);
  gemm128<<<(int)((ngw + 7) / 8), wgrp, 0, stream>>>(h1, W2t, b2, h16, h32);

  attn_logits<<<(int)((ngw + 7) / 8), wgrp, 0, stream>>>(h16, Ws1t, Ws2, logit);

  finalize<<<(N_NODES + 63) / 64, 128, 0, stream>>>(h32, logit, out);
}
